// WindowedAttentionWithRelPos_40261023432842
// MI455X (gfx1250) — compile-verified, matrix operand order checked by reading, not run on hardware
//
#include <hip/hip_runtime.h>
#include <hip/hip_bf16.h>
#include <math.h>

// ---------------------------------------------------------------------------
// WindowedAttentionWithRelPos for MI455X (gfx1250), wave32 + WMMA bf16.
// ---------------------------------------------------------------------------

typedef __attribute__((ext_vector_type(16))) __bf16 bf16x16;
typedef __attribute__((ext_vector_type(8)))  float  f32x8;

#define DIM    768
#define NHEAD  12
#define HDIM   64
#define SEQ    1024
#define BATCH  8
#define TOKENS 8192   // BATCH*SEQ
#define BNH    96     // BATCH*NHEAD

union FragU { bf16x16 v; uint4 q[2]; };

// Load a 16x32 bf16 fragment (rows striped over 16-lane halves, K-major per
// lane): lanes 0-15 hold row (lane), K {0..7,16..23}; lanes 16-31 hold row
// (lane-16), K {8..15,24..31}.  Same layout used for A and (column-of-B) frags.
__device__ __forceinline__ bf16x16 load_frag(const __bf16* base, int stride) {
  const int lane = threadIdx.x & 31;
  const __bf16* p = base + (lane & 15) * stride + ((lane & 16) ? 8 : 0);
  FragU u;
  u.q[0] = *(const uint4*)(p);
  u.q[1] = *(const uint4*)(p + 16);
  return u.v;
}

__device__ __forceinline__ f32x8 wmma_bf16(bf16x16 a, bf16x16 b, f32x8 c) {
  // (neg_a, A, neg_b, B, c_mod, C, reuse_a, reuse_b)
  return __builtin_amdgcn_wmma_f32_16x16x32_bf16(false, a, false, b,
                                                 (short)0, c, false, false);
}

// ---------------------------------------------------------------------------
// fp32 -> bf16 converters
// ---------------------------------------------------------------------------
__global__ __launch_bounds__(256) void cvt_kernel(const float* __restrict__ in,
                                                  __bf16* __restrict__ out, int n) {
  int i = blockIdx.x * 256 + threadIdx.x;
  if (i < n) out[i] = (__bf16)in[i];
}

// in: [Kd][Nd] fp32 row-major  ->  out: [Nd][Kd] bf16 (transposed)
__global__ __launch_bounds__(256) void cvt_t_kernel(const float* __restrict__ in,
                                                    __bf16* __restrict__ out,
                                                    int Kd, int Nd) {
  int i = blockIdx.x * 256 + threadIdx.x;
  if (i < Kd * Nd) {
    int k = i % Kd, n = i / Kd;
    out[i] = (__bf16)in[(size_t)k * Nd + n];
  }
}

// ---------------------------------------------------------------------------
// QKV GEMM: Xb[8192][768] @ WqkvT[2304][768]^T + bias -> scatter Q/K/Vt (bf16)
// Block: 128 threads (4 waves, 2x2), tile 128x128, K-step 32.
// ---------------------------------------------------------------------------
__global__ __launch_bounds__(128) void qkv_gemm_kernel(
    const __bf16* __restrict__ A, const __bf16* __restrict__ Bt,
    const float* __restrict__ bias,
    __bf16* __restrict__ Q, __bf16* __restrict__ Kb, __bf16* __restrict__ Vt) {
  __shared__ alignas(16) __bf16 sA[128][32];
  __shared__ alignas(16) __bf16 sB[128][32];
  const int tid  = threadIdx.x;
  const int lane = tid & 31;
  const int wv   = tid >> 5;
  const int wrow = (wv >> 1) * 64;
  const int wcol = (wv & 1) * 64;
  const int bx = blockIdx.x, by = blockIdx.y;

  const f32x8 fz = {};
  f32x8 acc[4][4];
#pragma unroll
  for (int i = 0; i < 4; ++i)
#pragma unroll
    for (int j = 0; j < 4; ++j) acc[i][j] = fz;

  const size_t rowA = (size_t)bx * 128 + tid;
  const size_t rowB = (size_t)by * 128 + tid;

  for (int k0 = 0; k0 < DIM; k0 += 32) {
    const uint4* ga = (const uint4*)(A  + rowA * DIM + k0);
    const uint4* gb = (const uint4*)(Bt + rowB * DIM + k0);
    uint4* la = (uint4*)(&sA[tid][0]);
    uint4* lb = (uint4*)(&sB[tid][0]);
#pragma unroll
    for (int u = 0; u < 4; ++u) { la[u] = ga[u]; lb[u] = gb[u]; }
    __syncthreads();
    bf16x16 af[4];
#pragma unroll
    for (int i = 0; i < 4; ++i) af[i] = load_frag(&sA[wrow + i * 16][0], 32);
#pragma unroll
    for (int j = 0; j < 4; ++j) {
      bf16x16 bfj = load_frag(&sB[wcol + j * 16][0], 32);
#pragma unroll
      for (int i = 0; i < 4; ++i) acc[i][j] = wmma_bf16(af[i], bfj, acc[i][j]);
    }
    __syncthreads();
  }

  // Epilogue: C layout -> scatter into Q[bn][s][d], K[bn][s][d], Vt[bn][d][s]
  const int halfr = (lane >> 4) * 8;
  const int cc = lane & 15;
#pragma unroll
  for (int i = 0; i < 4; ++i) {
#pragma unroll
    for (int j = 0; j < 4; ++j) {
#pragma unroll
      for (int r = 0; r < 8; ++r) {
        int grow = bx * 128 + wrow + i * 16 + r + halfr;  // token 0..8191
        int gcol = by * 128 + wcol + j * 16 + cc;         // 0..2303
        float v = acc[i][j][r] + bias[gcol];
        int b = grow >> 10, s = grow & 1023;
        int which = gcol / DIM, col = gcol % DIM;
        int h = col >> 6, d = col & 63;
        int bn = b * NHEAD + h;
        __bf16 bv = (__bf16)v;
        if (which == 0)      Q [((size_t)(bn * SEQ + s) << 6) + d] = bv;
        else if (which == 1) Kb[((size_t)(bn * SEQ + s) << 6) + d] = bv;
        else                 Vt[((size_t)(bn * HDIM + d) << 10) + s] = bv;
      }
    }
  }
}

// ---------------------------------------------------------------------------
// Decomposed rel-pos tables: relh/relw[bn][s][k] = sum_d Q[bn][s][d]*R[h-k+31][d]
// ---------------------------------------------------------------------------
__global__ __launch_bounds__(256) void relbias_kernel(
    const __bf16* __restrict__ Q, const float* __restrict__ rph,
    const float* __restrict__ rpw,
    float* __restrict__ relh, float* __restrict__ relw) {
  int idx = blockIdx.x * 256 + threadIdx.x;
  if (idx >= BNH * SEQ * 32) return;
  int k  = idx & 31;
  int s  = (idx >> 5) & (SEQ - 1);
  int bn = idx >> 15;
  int h = s >> 5, w = s & 31;
  const __bf16* q = Q + ((size_t)(bn * SEQ + s) << 6);
  const float* rh = rph + (size_t)(h - k + 31) * HDIM;
  const float* rw = rpw + (size_t)(w - k + 31) * HDIM;
  float ah = 0.f, aw = 0.f;
#pragma unroll 8
  for (int d = 0; d < HDIM; ++d) {
    float qv = (float)q[d];
    ah = fmaf(qv, rh[d], ah);
    aw = fmaf(qv, rw[d], aw);
  }
  relh[idx] = ah;
  relw[idx] = aw;
}

// ---------------------------------------------------------------------------
// Flash attention: one wave = 16 query rows; 32-key chunks; online softmax.
// Block 256 threads (8 waves) covers 128 query rows of one (b,head).
// ---------------------------------------------------------------------------
__global__ __launch_bounds__(256) void attn_kernel(
    const __bf16* __restrict__ Q, const __bf16* __restrict__ Kb,
    const __bf16* __restrict__ Vt,
    const float* __restrict__ relh, const float* __restrict__ relw,
    __bf16* __restrict__ AO) {
  __shared__ alignas(16) __bf16 sP[8][16][32];   // per-wave P tile
  const int bn   = blockIdx.x;
  const int wv   = threadIdx.x >> 5;
  const int lane = threadIdx.x & 31;
  const int s0   = blockIdx.y * 128 + wv * 16;
  const int halfr = (lane >> 4) * 8;
  const int cc = lane & 15;
  const float scale = 0.125f;  // 64^-0.5

  // Q fragments for rows s0..s0+15, d 0..63 (two K-chunks of 32)
  const __bf16* qbase = Q + ((size_t)(bn * SEQ + s0) << 6);
  bf16x16 qf[2];
  qf[0] = load_frag(qbase, HDIM);
  qf[1] = load_frag(qbase + 32, HDIM);

  const f32x8 fz = {};
  f32x8 o[4];
#pragma unroll
  for (int jd = 0; jd < 4; ++jd) o[jd] = fz;
  float m_run[8], l_run[8];
#pragma unroll
  for (int r = 0; r < 8; ++r) { m_run[r] = -1e30f; l_run[r] = 0.f; }

  for (int t0 = 0; t0 < SEQ; t0 += 32) {
    const int kh = t0 >> 5;  // constant over an aligned 32-key chunk
    f32x8 sc[2];
#pragma unroll
    for (int j = 0; j < 2; ++j) {
      const __bf16* kbase = Kb + ((size_t)(bn * SEQ + t0 + j * 16) << 6);
      f32x8 s = fz;
      s = wmma_bf16(qf[0], load_frag(kbase, HDIM), s);
      s = wmma_bf16(qf[1], load_frag(kbase + 32, HDIM), s);
      sc[j] = s;
    }
    // scale + rel-pos bias + online softmax (row = r + halfr of this tile)
#pragma unroll
    for (int r = 0; r < 8; ++r) {
      const int srow = s0 + r + halfr;
      const size_t rb = ((size_t)(bn * SEQ + srow)) << 5;
      float bh = relh[rb + kh];
      float v0 = fmaf(sc[0][r], scale, bh + relw[rb + cc]);
      float v1 = fmaf(sc[1][r], scale, bh + relw[rb + 16 + cc]);
      float mx = fmaxf(v0, v1);
#pragma unroll
      for (int m = 1; m < 16; m <<= 1) mx = fmaxf(mx, __shfl_xor(mx, m, 32));
      float mn = fmaxf(m_run[r], mx);
      float alpha = __expf(m_run[r] - mn);
      m_run[r] = mn;
      float p0 = __expf(v0 - mn);
      float p1 = __expf(v1 - mn);
      float rs = p0 + p1;
#pragma unroll
      for (int m = 1; m < 16; m <<= 1) rs += __shfl_xor(rs, m, 32);
      l_run[r] = l_run[r] * alpha + rs;
#pragma unroll
      for (int jd = 0; jd < 4; ++jd) o[jd][r] *= alpha;
      sP[wv][r + halfr][cc]      = (__bf16)p0;
      sP[wv][r + halfr][16 + cc] = (__bf16)p1;
    }
    // same-wave DS ops are in-order; the wait is a compiler fence + explicit
    // CDNA5 split-counter drain before re-reading the tile in A-layout.
    asm volatile("s_wait_dscnt 0" ::: "memory");
    bf16x16 pf = load_frag(&sP[wv][0][0], 32);
#pragma unroll
    for (int jd = 0; jd < 4; ++jd) {
      const __bf16* vb = Vt + ((size_t)(bn * HDIM + jd * 16) << 10) + t0;
      o[jd] = wmma_bf16(pf, load_frag(vb, SEQ), o[jd]);
    }
  }

  const int b = bn / NHEAD, head = bn % NHEAD;
#pragma unroll
  for (int r = 0; r < 8; ++r) {
    const int srow = s0 + r + halfr;
    const float inv = 1.0f / l_run[r];
    const size_t ob = (size_t)(b * SEQ + srow) * DIM + head * HDIM;
#pragma unroll
    for (int jd = 0; jd < 4; ++jd)
      AO[ob + jd * 16 + cc] = (__bf16)(o[jd][r] * inv);
  }
}

// ---------------------------------------------------------------------------
// Output projection: AO[8192][768] @ WprojT[768][768]^T + bias -> fp32 out
// ---------------------------------------------------------------------------
__global__ __launch_bounds__(128) void proj_gemm_kernel(
    const __bf16* __restrict__ A, const __bf16* __restrict__ Bt,
    const float* __restrict__ bias, float* __restrict__ Out) {
  __shared__ alignas(16) __bf16 sA[128][32];
  __shared__ alignas(16) __bf16 sB[128][32];
  const int tid  = threadIdx.x;
  const int lane = tid & 31;
  const int wv   = tid >> 5;
  const int wrow = (wv >> 1) * 64;
  const int wcol = (wv & 1) * 64;
  const int bx = blockIdx.x, by = blockIdx.y;

  const f32x8 fz = {};
  f32x8 acc[4][4];
#pragma unroll
  for (int i = 0; i < 4; ++i)
#pragma unroll
    for (int j = 0; j < 4; ++j) acc[i][j] = fz;

  const size_t rowA = (size_t)bx * 128 + tid;
  const size_t rowB = (size_t)by * 128 + tid;

  for (int k0 = 0; k0 < DIM; k0 += 32) {
    const uint4* ga = (const uint4*)(A  + rowA * DIM + k0);
    const uint4* gb = (const uint4*)(Bt + rowB * DIM + k0);
    uint4* la = (uint4*)(&sA[tid][0]);
    uint4* lb = (uint4*)(&sB[tid][0]);
#pragma unroll
    for (int u = 0; u < 4; ++u) { la[u] = ga[u]; lb[u] = gb[u]; }
    __syncthreads();
    bf16x16 af[4];
#pragma unroll
    for (int i = 0; i < 4; ++i) af[i] = load_frag(&sA[wrow + i * 16][0], 32);
#pragma unroll
    for (int j = 0; j < 4; ++j) {
      bf16x16 bfj = load_frag(&sB[wcol + j * 16][0], 32);
#pragma unroll
      for (int i = 0; i < 4; ++i) acc[i][j] = wmma_bf16(af[i], bfj, acc[i][j]);
    }
    __syncthreads();
  }

  const int halfr = (lane >> 4) * 8;
  const int cc = lane & 15;
#pragma unroll
  for (int i = 0; i < 4; ++i) {
#pragma unroll
    for (int j = 0; j < 4; ++j) {
#pragma unroll
      for (int r = 0; r < 8; ++r) {
        int grow = bx * 128 + wrow + i * 16 + r + halfr;
        int gcol = by * 128 + wcol + j * 16 + cc;
        Out[(size_t)grow * DIM + gcol] = acc[i][j][r] + bias[gcol];
      }
    }
  }
}

// ---------------------------------------------------------------------------
// Host-side launch
// ---------------------------------------------------------------------------
extern "C" void kernel_launch(void* const* d_in, const int* in_sizes, int n_in,
                              void* d_out, int out_size, void* d_ws, size_t ws_size,
                              hipStream_t stream) {
  (void)in_sizes; (void)n_in; (void)out_size; (void)ws_size;
  const float* x     = (const float*)d_in[0];
  const float* wqkv  = (const float*)d_in[1];
  const float* bqkv  = (const float*)d_in[2];
  const float* wproj = (const float*)d_in[3];
  const float* bproj = (const float*)d_in[4];
  const float* rph   = (const float*)d_in[5];
  const float* rpw   = (const float*)d_in[6];
  float* out = (float*)d_out;

  char* p = (char*)d_ws;
  size_t off = 0;
  auto take = [&](size_t bytes) -> void* {
    void* r = p + off;
    off += (bytes + 255) & ~(size_t)255;
    return r;
  };
  __bf16* Xb     = (__bf16*)take((size_t)TOKENS * DIM * 2);
  __bf16* WqkvT  = (__bf16*)take((size_t)3 * DIM * DIM * 2);
  __bf16* WprojT = (__bf16*)take((size_t)DIM * DIM * 2);
  __bf16* Qb     = (__bf16*)take((size_t)BNH * SEQ * HDIM * 2);
  __bf16* Kb     = (__bf16*)take((size_t)BNH * SEQ * HDIM * 2);
  __bf16* Vt     = (__bf16*)take((size_t)BNH * HDIM * SEQ * 2);
  float*  relh   = (float*) take((size_t)BNH * SEQ * 32 * 4);
  float*  relw   = (float*) take((size_t)BNH * SEQ * 32 * 4);
  __bf16* AO     = (__bf16*)take((size_t)TOKENS * DIM * 2);

  int n1 = TOKENS * DIM;
  cvt_kernel<<<(n1 + 255) / 256, 256, 0, stream>>>(x, Xb, n1);
  int n2 = 3 * DIM * DIM;
  cvt_t_kernel<<<(n2 + 255) / 256, 256, 0, stream>>>(wqkv, WqkvT, DIM, 3 * DIM);
  int n3 = DIM * DIM;
  cvt_t_kernel<<<(n3 + 255) / 256, 256, 0, stream>>>(wproj, WprojT, DIM, DIM);

  qkv_gemm_kernel<<<dim3(TOKENS / 128, (3 * DIM) / 128), 128, 0, stream>>>(
      Xb, WqkvT, bqkv, Qb, Kb, Vt);

  int n4 = BNH * SEQ * 32;
  relbias_kernel<<<(n4 + 255) / 256, 256, 0, stream>>>(Qb, rph, rpw, relh, relw);

  attn_kernel<<<dim3(BNH, SEQ / 128), 256, 0, stream>>>(Qb, Kb, Vt, relh, relw, AO);

  proj_gemm_kernel<<<dim3(TOKENS / 128, DIM / 128), 128, 0, stream>>>(
      AO, WprojT, bproj, out);
}
